// HierarchicalEEGTransformer_30760555774505
// MI455X (gfx1250) — matrix operand order repaired and checked by reading, not compiled, not run on hardware
//
#include <hip/hip_runtime.h>

typedef __attribute__((ext_vector_type(16))) _Float16 v16h;
typedef __attribute__((ext_vector_type(8)))  _Float16 v8h;
typedef __attribute__((ext_vector_type(8)))  float    v8f;
typedef __attribute__((ext_vector_type(4))) unsigned int u32x4;
typedef __attribute__((ext_vector_type(8))) int  i32x8;
typedef __attribute__((ext_vector_type(4))) int  i32x4;

#define D_    768
#define H_    12
#define DH_   64
#define W_    128
#define NW_   8
#define L_    2
#define V_    1024
#define B_    2
#define M_    (B_*NW_*W_)   /* 2048 rows */
#define DFF_  (4*D_)        /* 3072 */

static __device__ __forceinline__ v8f wmma_f16(const v16h& a, const v16h& b, const v8f& c) {
  return __builtin_amdgcn_wmma_f32_16x16x32_f16(false, a, false, b, (short)0, c, false, false);
}

// ---------------- f32 -> f16 conversion ----------------
__global__ __launch_bounds__(256) void cvt_f32_f16(const float* __restrict__ in,
                                                   _Float16* __restrict__ out, int n) {
  int i = blockIdx.x * 256 + threadIdx.x;
  if (i < n) out[i] = (_Float16)in[i];
}

// ---------------- embedding gather ----------------
__global__ __launch_bounds__(256) void embed_kernel(const int* __restrict__ x,
                                                    const float* __restrict__ emb,
                                                    const float* __restrict__ wpos,
                                                    const float* __restrict__ tpos,
                                                    float* __restrict__ h) {
  int row = blockIdx.x;            // 0..2047
  int b   = row >> 10;
  int rem = row & 1023;
  int wi  = rem >> 7;
  int t   = rem & 127;
  int tok = x[b * NW_ * (W_ + 1) + wi * (W_ + 1) + t];
  const float* e  = emb  + (size_t)tok * D_;
  const float* wp = wpos + (size_t)wi  * D_;
  const float* tp = tpos + (size_t)t   * D_;
  float* o = h + (size_t)row * D_;
  for (int d = threadIdx.x; d < D_; d += 256)
    o[d] = e[d] + wp[d] + tp[d];
}

// ---------------- layernorm (f32 in, f16 out) ----------------
__global__ __launch_bounds__(256) void ln_kernel(const float* __restrict__ in,
                                                 _Float16* __restrict__ out,
                                                 const float* __restrict__ g,
                                                 const float* __restrict__ bta) {
  int row = blockIdx.x;
  const float* p = in + (size_t)row * D_;
  float v0 = p[threadIdx.x], v1 = p[threadIdx.x + 256], v2 = p[threadIdx.x + 512];
  float s = v0 + v1 + v2;
  float q = v0 * v0 + v1 * v1 + v2 * v2;
  for (int m = 1; m < 32; m <<= 1) {
    s += __shfl_xor(s, m, 32);
    q += __shfl_xor(q, m, 32);
  }
  __shared__ float sh_s[8], sh_q[8];
  int wid = threadIdx.x >> 5;
  if ((threadIdx.x & 31) == 0) { sh_s[wid] = s; sh_q[wid] = q; }
  __syncthreads();
  s = 0.f; q = 0.f;
  for (int i = 0; i < 8; ++i) { s += sh_s[i]; q += sh_q[i]; }
  float mean = s * (1.0f / D_);
  float var  = q * (1.0f / D_) - mean * mean;
  float inv  = rsqrtf(var + 1e-5f);
  _Float16* o = out + (size_t)row * D_;
  int d = threadIdx.x;
  o[d]       = (_Float16)((v0 - mean) * inv * g[d]       + bta[d]);
  o[d + 256] = (_Float16)((v1 - mean) * inv * g[d + 256] + bta[d + 256]);
  o[d + 512] = (_Float16)((v2 - mean) * inv * g[d + 512] + bta[d + 512]);
}

// ---------------- WMMA GEMM: out = act(A[M,K] @ W[K,N] + bias) (+res) ----------------
// Block: 256 threads = 8 waves. Block tile 128(M) x 64(N). Wave: 16x64 (4 accumulators).
// Double-buffered LDS B tile -> single barrier per K step.
__global__ __launch_bounds__(256) void gemm_kernel(const _Float16* __restrict__ A,
                                                   const _Float16* __restrict__ Wt,
                                                   const float* __restrict__ bias,
                                                   const float* res,
                                                   float* outF, _Float16* outH,
                                                   int K, int N, int do_gelu) {
  __shared__ _Float16 BtT[2][64 * 40];        // B tile transposed [n][k], padded stride
  const int lane  = threadIdx.x & 31;
  const int wid   = threadIdx.x >> 5;
  const int l15   = lane & 15;
  const int hi16  = (lane < 16) ? 0 : 1;
  const int koffA = hi16 ? 16 : 0;
  const int koffB = hi16 ? 16 : 0;
  const int row0  = blockIdx.x * 128 + wid * 16;
  const int n0    = blockIdx.y * 64;
  const int m     = row0 + l15;
  const int sk_row = threadIdx.x >> 3;        // 0..31  (k within tile)
  const int sk_col = (threadIdx.x & 7) * 8;   // 0..56  (n within tile)

  const v8f zero = {0.f, 0.f, 0.f, 0.f, 0.f, 0.f, 0.f, 0.f};
  v8f acc[4] = {zero, zero, zero, zero};

  int buf = 0;
  for (int kk = 0; kk < K; kk += 32) {
    // stage W[kk..kk+32, n0..n0+64] transposed into the current LDS buffer
    v8h wrow = *reinterpret_cast<const v8h*>(Wt + (size_t)(kk + sk_row) * N + n0 + sk_col);
    // A fragment global load issued early (no LDS dependency)
    const _Float16* ap = A + (size_t)m * K + kk + koffA;
    v8h alo = *reinterpret_cast<const v8h*>(ap);
    v8h ahi = *reinterpret_cast<const v8h*>(ap + 8);
    if (kk + 32 < K)
      __builtin_prefetch(Wt + (size_t)(kk + 32 + sk_row) * N + n0 + sk_col, 0, 1);

    _Float16* bdst = &BtT[buf][0];
    #pragma unroll
    for (int i = 0; i < 8; ++i) bdst[(sk_col + i) * 40 + sk_row] = wrow[i];
    __syncthreads();

    v16h a;
    #pragma unroll
    for (int i = 0; i < 8; ++i) { a[i] = alo[i]; a[i + 8] = ahi[i]; }

    #pragma unroll
    for (int t = 0; t < 4; ++t) {
      const _Float16* bp = &BtT[buf][(t * 16 + l15) * 40 + koffB];
      v8h blo = *reinterpret_cast<const v8h*>(bp);
      v8h bhi = *reinterpret_cast<const v8h*>(bp + 8);
      v16h bf;
      #pragma unroll
      for (int i = 0; i < 8; ++i) { bf[i] = blo[i]; bf[i + 8] = bhi[i]; }
      acc[t] = wmma_f16(a, bf, acc[t]);
    }
    buf ^= 1;
  }

  // epilogue
  #pragma unroll
  for (int t = 0; t < 4; ++t) {
    #pragma unroll
    for (int r = 0; r < 8; ++r) {
      int rr = row0 + r + hi16 * 8;
      int cc = n0 + t * 16 + l15;
      float v = acc[t][r] + bias[cc];
      if (do_gelu) v = 0.5f * v * (1.0f + erff(v * 0.70710678118f));
      if (res)  v += res[(size_t)rr * N + cc];
      if (outF) outF[(size_t)rr * N + cc] = v;
      else      outH[(size_t)rr * N + cc] = (_Float16)v;
    }
  }
}

// ---------------- block-causal windowed attention ----------------
// grid (NW, H, B): one workgroup per (query window i, head, batch).
// Per j <= i: TDM DMAs the V_j tile (128x64 f16, HW-padded LDS rows) while the
// waves compute S = Q_i K_j^T with WMMA; register softmax (shfl within 16-lane
// halves); A -> LDS f16; ctx += A @ V_j with WMMA.
__global__ __launch_bounds__(256) void attn_kernel(const _Float16* __restrict__ Q,
                                                   const _Float16* __restrict__ Km,
                                                   const _Float16* __restrict__ Vm,
                                                   _Float16* __restrict__ ctx) {
  __shared__ _Float16 A16[128 * 136];
  __shared__ _Float16 Vt[128 * 72];
  const int i    = blockIdx.x;
  const int head = blockIdx.y;
  const int b    = blockIdx.z;
  const int lane = threadIdx.x & 31;
  const int wid  = threadIdx.x >> 5;
  const int l15  = lane & 15;
  const int hi16 = (lane < 16) ? 0 : 1;
  const int koffA = hi16 ? 16 : 0;
  const int koffB = hi16 ? 16 : 0;
  const int tbase = b * (NW_ * W_) + i * W_;
  const int hc    = head * DH_;
  const int m     = tbase + wid * 16 + l15;   // global query row for this lane
  const float scale = 0.125f;                 // 1/sqrt(64)

  const v8f zero = {0.f, 0.f, 0.f, 0.f, 0.f, 0.f, 0.f, 0.f};
  v8f c[4] = {zero, zero, zero, zero};

  for (int j = 0; j <= i; ++j) {
    const int jb = b * (NW_ * W_) + j * W_;
    __syncthreads();   // previous iteration's readers of Vt/A16 are done

    // ---- TDM: async DMA V_j tile [128 x 64] f16 into LDS, row pad 64->72 halves
    if (wid == 0) {
      unsigned long long ga =
          (unsigned long long)(const void*)(Vm + (size_t)jb * D_ + hc);
      unsigned int ldsoff = (unsigned int)(unsigned long long)(const void*)&Vt[0];
      u32x4 g0;
      g0[0] = 1u;                                   // count=1, user descriptor
      g0[1] = ldsoff;                               // lds_addr (bytes)
      g0[2] = (unsigned int)(ga & 0xffffffffull);   // global_addr[31:0]
      g0[3] = (unsigned int)((ga >> 32) & 0x01fffffful) | (2u << 30); // addr[56:32], type=2
      i32x8 g1;
      g1[0] = (int)((1u << 16)  |   // data_size = 1 -> 2 bytes
                    (1u << 20)  |   // pad_enable
                    (4u << 22)  |   // pad_interval: 32 dwords (one 64-half row)
                    (3u << 25));    // pad_amount: 4 dwords (8 halves) -> stride 72
      g1[1] = (int)(64u  << 16);    // tensor_dim0 = 64 elements
      g1[2] = (int)(128u << 16);    // dim0 hi = 0 ; tensor_dim1 = 128 rows
      g1[3] = (int)(64u  << 16);    // dim1 hi = 0 ; tile_dim0 = 64
      g1[4] = (int)128u;            // tile_dim1 = 128, tile_dim2 = 0
      g1[5] = (int)768u;            // tensor_dim0_stride = 768 elements
      g1[6] = 0;
      g1[7] = 0;
      i32x4 gz4 = {0, 0, 0, 0};
      i32x8 gz8 = {0, 0, 0, 0, 0, 0, 0, 0};
      __builtin_amdgcn_tensor_load_to_lds(g0, g1, gz4, gz4, gz8, 0);
    }

    // ---- S = Q_i K_j^T : each wave does a 16x128 strip (overlaps the DMA) ----
    v8f s[8] = {zero, zero, zero, zero, zero, zero, zero, zero};
    for (int kk = 0; kk < DH_; kk += 32) {
      const _Float16* qp = Q + (size_t)m * D_ + hc + kk + koffA;
      v8h qlo = *reinterpret_cast<const v8h*>(qp);
      v8h qhi = *reinterpret_cast<const v8h*>(qp + 8);
      v16h a;
      #pragma unroll
      for (int xi = 0; xi < 8; ++xi) { a[xi] = qlo[xi]; a[xi + 8] = qhi[xi]; }
      #pragma unroll
      for (int t = 0; t < 8; ++t) {
        const int kt = jb + t * 16 + l15;
        const _Float16* kp = Km + (size_t)kt * D_ + hc + kk + koffB;
        v8h klo = *reinterpret_cast<const v8h*>(kp);
        v8h khi = *reinterpret_cast<const v8h*>(kp + 8);
        v16h bf;
        #pragma unroll
        for (int xi = 0; xi < 8; ++xi) { bf[xi] = klo[xi]; bf[xi + 8] = khi[xi]; }
        s[t] = wmma_f16(a, bf, s[t]);
      }
    }

    // ---- softmax per row over 128 keys (register + shfl), write A16 f16 ----
    #pragma unroll
    for (int r = 0; r < 8; ++r) {
      float mx = -3.0e38f;
      #pragma unroll
      for (int t = 0; t < 8; ++t) mx = fmaxf(mx, s[t][r]);
      for (int msk = 1; msk < 16; msk <<= 1) mx = fmaxf(mx, __shfl_xor(mx, msk, 32));
      float e[8];
      float sum = 0.f;
      #pragma unroll
      for (int t = 0; t < 8; ++t) { e[t] = expf((s[t][r] - mx) * scale); sum += e[t]; }
      for (int msk = 1; msk < 16; msk <<= 1) sum += __shfl_xor(sum, msk, 32);
      float rinv = 1.0f / sum;
      int rr = wid * 16 + r + hi16 * 8;
      #pragma unroll
      for (int t = 0; t < 8; ++t)
        A16[rr * 136 + t * 16 + l15] = (_Float16)(e[t] * rinv);
    }

    if (wid == 0) __builtin_amdgcn_s_wait_tensorcnt(0);  // V tile landed
    __syncthreads();

    // ---- ctx += A @ V_j : wave does 16x64, K-dim = 128 keys ----
    #pragma unroll
    for (int kk4 = 0; kk4 < 4; ++kk4) {
      const int kk = kk4 * 32;
      const _Float16* ap = &A16[(wid * 16 + l15) * 136 + kk + koffA];
      v8h alo = *reinterpret_cast<const v8h*>(ap);
      v8h ahi = *reinterpret_cast<const v8h*>(ap + 8);
      v16h a;
      #pragma unroll
      for (int xi = 0; xi < 8; ++xi) { a[xi] = alo[xi]; a[xi + 8] = ahi[xi]; }
      #pragma unroll
      for (int t = 0; t < 4; ++t) {
        v16h bf;
        #pragma unroll
        for (int s16 = 0; s16 < 16; ++s16)
          bf[s16] = Vt[(kk + koffB + s16) * 72 + t * 16 + l15];
        c[t] = wmma_f16(a, bf, c[t]);
      }
    }
  }

  // write ctx (f16) for this (b, i, head)
  #pragma unroll
  for (int t = 0; t < 4; ++t) {
    #pragma unroll
    for (int r = 0; r < 8; ++r) {
      int rr = tbase + wid * 16 + r + hi16 * 8;
      int cc = hc + t * 16 + l15;
      ctx[(size_t)rr * D_ + cc] = (_Float16)c[t][r];
    }
  }
}

// ---------------- host orchestration ----------------
extern "C" void kernel_launch(void* const* d_in, const int* in_sizes, int n_in,
                              void* d_out, int out_size, void* d_ws, size_t ws_size,
                              hipStream_t stream) {
  const int*   x    = (const int*)d_in[0];
  const float* emb  = (const float*)d_in[1];
  const float* wpos = (const float*)d_in[2];
  const float* tpos = (const float*)d_in[3];
  const float* Wq   = (const float*)d_in[4];
  const float* bq   = (const float*)d_in[5];
  const float* Wk   = (const float*)d_in[6];
  const float* bk   = (const float*)d_in[7];
  const float* Wv   = (const float*)d_in[8];
  const float* bv   = (const float*)d_in[9];
  const float* Wo   = (const float*)d_in[10];
  const float* bo   = (const float*)d_in[11];
  const float* ln1g = (const float*)d_in[12];
  const float* ln1b = (const float*)d_in[13];
  const float* W1   = (const float*)d_in[14];
  const float* b1   = (const float*)d_in[15];
  const float* W2   = (const float*)d_in[16];
  const float* b2   = (const float*)d_in[17];
  const float* ln2g = (const float*)d_in[18];
  const float* ln2b = (const float*)d_in[19];
  const float* lnfg = (const float*)d_in[20];
  const float* lnfb = (const float*)d_in[21];
  const float* Wout = (const float*)d_in[22];
  const float* bout = (const float*)d_in[23];
  (void)in_sizes; (void)n_in; (void)out_size; (void)ws_size;

  char* ws = (char*)d_ws;
  size_t off = 0;
  auto alloc = [&](size_t bytes) -> char* {
    char* p = ws + off;
    off += (bytes + 255) & ~(size_t)255;
    return p;
  };
  float*    h    = (float*)   alloc((size_t)M_ * D_ * 4);
  _Float16* z    = (_Float16*)alloc((size_t)M_ * D_ * 2);
  _Float16* qf   = (_Float16*)alloc((size_t)M_ * D_ * 2);
  _Float16* kf   = (_Float16*)alloc((size_t)M_ * D_ * 2);
  _Float16* vf   = (_Float16*)alloc((size_t)M_ * D_ * 2);
  _Float16* ctxf = (_Float16*)alloc((size_t)M_ * D_ * 2);
  _Float16* ff1f = (_Float16*)alloc((size_t)M_ * DFF_ * 2);
  _Float16* wq16 = (_Float16*)alloc((size_t)L_ * D_ * D_ * 2);
  _Float16* wk16 = (_Float16*)alloc((size_t)L_ * D_ * D_ * 2);
  _Float16* wv16 = (_Float16*)alloc((size_t)L_ * D_ * D_ * 2);
  _Float16* wo16 = (_Float16*)alloc((size_t)L_ * D_ * D_ * 2);
  _Float16* w116 = (_Float16*)alloc((size_t)L_ * D_ * DFF_ * 2);
  _Float16* w216 = (_Float16*)alloc((size_t)L_ * DFF_ * D_ * 2);
  _Float16* wout16 = (_Float16*)alloc((size_t)D_ * V_ * 2);

  auto cvt = [&](const float* src, _Float16* dst, size_t n) {
    cvt_f32_f16<<<dim3((unsigned)((n + 255) / 256)), 256, 0, stream>>>(src, dst, (int)n);
  };
  auto gemm = [&](const _Float16* Aa, const _Float16* Ww, const float* bb,
                  const float* rr, float* oF, _Float16* oH, int K, int N, int gelu) {
    gemm_kernel<<<dim3(M_ / 128, N / 64), 256, 0, stream>>>(Aa, Ww, bb, rr, oF, oH, K, N, gelu);
  };
  auto lnorm = [&](const float* in, _Float16* out, const float* g, const float* b) {
    ln_kernel<<<dim3(M_), 256, 0, stream>>>(in, out, g, b);
  };

  // weight conversions (f32 -> f16)
  cvt(Wq, wq16, (size_t)L_ * D_ * D_);
  cvt(Wk, wk16, (size_t)L_ * D_ * D_);
  cvt(Wv, wv16, (size_t)L_ * D_ * D_);
  cvt(Wo, wo16, (size_t)L_ * D_ * D_);
  cvt(W1, w116, (size_t)L_ * D_ * DFF_);
  cvt(W2, w216, (size_t)L_ * DFF_ * D_);
  cvt(Wout, wout16, (size_t)D_ * V_);

  // embedding
  embed_kernel<<<dim3(M_), 256, 0, stream>>>(x, emb, wpos, tpos, h);

  for (int l = 0; l < L_; ++l) {
    const size_t wOff  = (size_t)l * D_ * D_;
    const size_t w1Off = (size_t)l * D_ * DFF_;
    const size_t w2Off = (size_t)l * DFF_ * D_;

    lnorm(h, z, ln1g + l * D_, ln1b + l * D_);
    gemm(z, wq16 + wOff, bq + l * D_, nullptr, nullptr, qf, D_, D_, 0);
    gemm(z, wk16 + wOff, bk + l * D_, nullptr, nullptr, kf, D_, D_, 0);
    gemm(z, wv16 + wOff, bv + l * D_, nullptr, nullptr, vf, D_, D_, 0);

    attn_kernel<<<dim3(NW_, H_, B_), 256, 0, stream>>>(qf, kf, vf, ctxf);

    // h = h + ctx @ Wo + bo   (residual fused)
    gemm(ctxf, wo16 + wOff, bo + l * D_, h, h, nullptr, D_, D_, 0);

    lnorm(h, z, ln2g + l * D_, ln2b + l * D_);
    gemm(z, w116 + w1Off, b1 + l * DFF_, nullptr, nullptr, ff1f, D_, DFF_, 1);  // gelu
    gemm(ff1f, w216 + w2Off, b2 + l * D_, h, h, nullptr, DFF_, D_, 0);          // residual
  }

  lnorm(h, z, lnfg, lnfb);
  gemm(z, wout16, bout, nullptr, (float*)d_out, nullptr, D_, V_, 0);
}
